// ResNetBase_12713103197219
// MI455X (gfx1250) — matrix operand order repaired and checked by reading, not compiled, not run on hardware
//
#include <hip/hip_runtime.h>
#include <hip/hip_bf16.h>
#include <stdint.h>

typedef _Float16 h16;
typedef __attribute__((ext_vector_type(8)))  _Float16 v8h;
typedef __attribute__((ext_vector_type(16))) _Float16 v16h;
typedef __attribute__((ext_vector_type(8)))  float    v8f;

// ---------------- reference-model constants ----------------
#define RN_N0 300000
#define RN_N1 150000
#define RN_N2 75000
#define RN_N5 2000
#define RN_B  8
#define RN_CIN 3
#define RN_C0 64
#define RN_OUT 20
#define RN_K1 125
#define RN_K3 27
#define RN_M1 8000
static const int RN_NL[4] = {40000, 20000, 10000, 5000};
static const int RN_PL[4] = {64, 128, 256, 512};

static inline int ceil_div_ll(long long a, long long b) { return (int)((a + b - 1) / b); }

// ---------------- kernels ----------------

// Pack fp32 weights [K, Cin, Cout] into the per-lane WMMA B-operand layout:
// P[k][nt][ci][lane][h] = (f16) W[k][ci*32 + (h>>3)*16 + (lane>>4)*8 + (h&7)][nt*16 + (lane&15)]
// (zero-padded for kk >= Cin). Each lane's B fragment becomes 32 contiguous bytes.
__global__ __launch_bounds__(256) void k_pack_w(const float* __restrict__ W, h16* __restrict__ P,
                                                int K, int Cin, int Cout, int CC /* CinPad/32 */)
{
    const long long total = (long long)K * (Cout >> 4) * CC * 512;
    long long i = blockIdx.x * 256LL + threadIdx.x;
    if (i >= total) return;
    const int h    = (int)(i & 15);
    const int lane = (int)((i >> 4) & 31);
    long long r    = i >> 9;
    const int ci   = (int)(r % CC);  r /= CC;
    const int NT   = Cout >> 4;
    const int nt   = (int)(r % NT);
    const int k    = (int)(r / NT);
    const int kk   = ci * 32 + ((h >> 3) << 4) + ((lane >> 4) << 3) + (h & 7);
    const int col  = (nt << 4) + (lane & 15);
    P[i] = (kk < Cin) ? (h16)W[((size_t)k * Cin + kk) * Cout + col] : (h16)0.f;
}

// Generic sparse conv: out[iout[k,m], :] += f16in[iin[k,m], :] @ W[k]
// f16 A/B fragments, fp32 accumulation. One wave = one 16x16 (M x Cout) tile of one offset k.
// Cin must be a multiple of 32 (conv1 input is zero-padded to 32).
__global__ __launch_bounds__(256) void k_sparse_conv_wmma(
    const h16* __restrict__ fin,      // [Nin, Cin] f16 (row stride = Cin)
    const h16* __restrict__ Wp,       // packed B operands (see k_pack_w)
    const int* __restrict__ iin,      // [K, M]
    const int* __restrict__ iout,     // [K, M] or nullptr (identity)
    float* __restrict__ out,          // [Nout, Cout] fp32 accumulator (pre-zeroed)
    int K, int M, int Cin, int Cout)
{
    const int lane = threadIdx.x & 31;
    const int wid  = blockIdx.x * (blockDim.x >> 5) + (threadIdx.x >> 5);
    const int NT = Cout >> 4;
    const int MT = (M + 15) >> 4;
    const int total = K * MT * NT;
    if (wid >= total) return;                 // wave-uniform: EXEC stays all-ones for WMMA

    const int k  = wid / (MT * NT);
    const int rm = wid - k * (MT * NT);
    const int mt = rm / NT;
    const int nt = rm - mt * NT;
    const int m0 = mt << 4, n0 = nt << 4;
    const int r  = lane & 15;                 // A row within tile / D column (N)
    const int hh = lane >> 4;                 // lane half

    const int mr = m0 + r;
    const int gr = iin[k * M + (mr < M ? mr : (M - 1))];
    const int CC = Cin >> 5;
    const h16* __restrict__ arow = fin + (size_t)gr * Cin + (hh << 3);
    const h16* __restrict__ bp   = Wp + ((((size_t)k * NT + nt) * CC) * 32 + lane) * 16;

    v8f acc = {};
    auto step = [&](int ci) {
        const v8h alo = *(const v8h*)(arow + (ci << 5));        // K = hh*8 + 0..7
        const v8h ahi = *(const v8h*)(arow + (ci << 5) + 16);   // K = 16 + hh*8 + 0..7
        const v16h a = __builtin_shufflevector(alo, ahi, 0,1,2,3,4,5,6,7,8,9,10,11,12,13,14,15);
        const v16h b = *(const v16h*)(bp + (size_t)ci * 512);   // 32B contiguous
        acc = __builtin_amdgcn_wmma_f32_16x16x32_f16(false, a, false, b,
                                                     (short)0, acc, false, false);
    };
    int ci = 0;
    for (; ci + 2 <= CC; ci += 2) { step(ci); step(ci + 1); }   // 2x unroll; both WMMAs chain through acc
    if (ci < CC) step(ci);

    #pragma unroll
    for (int v = 0; v < 8; ++v) {             // D: (M = v + 8*half, N = r)
        const int mm = m0 + v + (hh << 3);
        if (mm < M) {
            const int orow = iout ? iout[k * M + mm] : mm;
            atomicAdd(out + (size_t)orow * Cout + n0 + r, acc[v]);
        }
    }
}

__global__ __launch_bounds__(256) void k_zero(float* p, long long n) {
    long long i = blockIdx.x * 256LL + threadIdx.x;
    if (i < n) p[i] = 0.f;
}

// x [N0,3] fp32 -> padded [N0,32] f16 (zero tail) so conv1 runs the guard-free path
__global__ __launch_bounds__(256) void k_cvt_pad(const float* __restrict__ x, h16* __restrict__ y,
                                                 long long n /* N0*32 */)
{
    long long i = blockIdx.x * 256LL + threadIdx.x;
    if (i >= n) return;
    const int c = (int)(i & 31);
    const long long row = i >> 5;
    y[i] = (c < RN_CIN) ? (h16)x[row * RN_CIN + c] : (h16)0.f;
}

// per-channel sum / sumsq via LDS partials (C is a power of two)
__global__ __launch_bounds__(256) void k_bn_stats(const float* __restrict__ t,
                                                  float* __restrict__ sums, float* __restrict__ sqs,
                                                  long long total, int C)
{
    extern __shared__ float sm[];             // [2*C]
    float* s0 = sm; float* s1 = sm + C;
    for (int i = threadIdx.x; i < 2 * C; i += blockDim.x) sm[i] = 0.f;
    __syncthreads();
    const int cmask = C - 1;
    for (long long i = blockIdx.x * 256LL + threadIdx.x; i < total; i += (long long)gridDim.x * 256LL) {
        const int c = (int)(i & cmask);
        const float v = t[i];
        atomicAdd(&s0[c], v);
        atomicAdd(&s1[c], v * v);
    }
    __syncthreads();
    for (int c = threadIdx.x; c < C; c += blockDim.x) {
        atomicAdd(&sums[c], s0[c]);
        atomicAdd(&sqs[c],  s1[c]);
    }
}

__global__ __launch_bounds__(256) void k_bn_fin(const float* sums, const float* sqs,
                                                const float* g, const float* b,
                                                float* scale, float* shift, int N, int C)
{
    int c = blockIdx.x * 256 + threadIdx.x;
    if (c < C) {
        const float invN = 1.f / (float)N;
        const float mu  = sums[c] * invN;
        const float var = sqs[c] * invN - mu * mu;
        const float sc  = g[c] * rsqrtf(var + 1e-5f);
        scale[c] = sc;
        shift[c] = b[c] - sc * mu;
    }
}

// v = bnA(t) [+ res or bnB(res)], ReLU; writes fp32 + optional f16 shadow
__global__ __launch_bounds__(256) void k_bn_apply(const float* __restrict__ t,
                                                  const float* __restrict__ sc, const float* __restrict__ sh,
                                                  const float* __restrict__ res,
                                                  const float* __restrict__ scB, const float* __restrict__ shB,
                                                  float* __restrict__ o32, h16* __restrict__ o16,
                                                  long long n, int cmask)
{
    long long i = blockIdx.x * 256LL + threadIdx.x;
    if (i >= n) return;
    const int c = (int)(i & cmask);
    float v = fmaf(t[i], sc[c], sh[c]);
    if (res) {
        float rv = res[i];
        if (scB) rv = fmaf(rv, scB[c], shB[c]);
        v += rv;
    }
    v = fmaxf(v, 0.f);
    o32[i] = v;
    if (o16) o16[i] = (h16)v;
}

__global__ __launch_bounds__(256) void k_pool_scatter(const float* __restrict__ f,
                                                      const int* __restrict__ iin, const int* __restrict__ iout,
                                                      float* __restrict__ sum, float* __restrict__ cnt,
                                                      long long n /* P*C */, int cshift)
{
    long long i = blockIdx.x * 256LL + threadIdx.x;
    if (i >= n) return;
    const long long p = i >> cshift;
    const int c = (int)(i & ((1 << cshift) - 1));
    const int ir = iin[p], orow = iout[p];
    atomicAdd(&sum[((size_t)orow << cshift) + c], f[((size_t)ir << cshift) + c]);
    if (c == 0) atomicAdd(&cnt[orow], 1.f);
}

__global__ __launch_bounds__(256) void k_pool_fin(float* __restrict__ s, const float* __restrict__ cnt,
                                                  h16* __restrict__ o16, long long n, int cshift)
{
    long long i = blockIdx.x * 256LL + threadIdx.x;
    if (i >= n) return;
    const float v = s[i] / fmaxf(cnt[i >> cshift], 1.f);
    s[i] = v;
    o16[i] = (h16)v;
}

// values are post-ReLU (>=0): int compare == float compare; g pre-zeroed
__global__ __launch_bounds__(256) void k_seg_max(const float* __restrict__ f, const int* __restrict__ seg,
                                                 float* __restrict__ g, long long n, int cshift)
{
    long long i = blockIdx.x * 256LL + threadIdx.x;
    if (i >= n) return;
    const int row = (int)(i >> cshift);
    const int c   = (int)(i & ((1 << cshift) - 1));
    atomicMax((int*)&g[((size_t)seg[row] << cshift) + c], __float_as_int(f[i]));
}

__global__ __launch_bounds__(256) void k_fc(const float* __restrict__ g, const float* __restrict__ w,
                                            const float* __restrict__ bias, float* __restrict__ out,
                                            int Bn, int Ci, int Co)
{
    int i = blockIdx.x * 256 + threadIdx.x;
    if (i >= Bn * Co) return;
    const int bb = i / Co, o = i % Co;
    float acc = bias[o];
    for (int c = 0; c < Ci; ++c) acc = fmaf(g[bb * Ci + c], w[c * Co + o], acc);
    out[i] = acc;
}

// ---------------- host helpers ----------------
static void h_zero(float* p, long long n, hipStream_t s) {
    k_zero<<<dim3(ceil_div_ll(n, 256)), dim3(256), 0, s>>>(p, n);
}

static void h_conv(const h16* fin, const h16* Wp, const int* iin, const int* iout,
                   float* out, int K, int M, int Cin, int Cout, int Nout, hipStream_t s)
{
    h_zero(out, (long long)Nout * Cout, s);
    const long long tiles = (long long)K * ((M + 15) / 16) * (Cout / 16);
    k_sparse_conv_wmma<<<dim3(ceil_div_ll(tiles, 8)), dim3(256), 0, s>>>(
        fin, Wp, iin, iout, out, K, M, Cin, Cout);
}

static void h_bn(const float* t, const float* g, const float* b, int N, int C,
                 float* sums /*2*C scratch*/, float* scale, float* shift, hipStream_t s)
{
    h_zero(sums, 2 * C, s);
    k_bn_stats<<<dim3(768), dim3(256), 2 * C * sizeof(float), s>>>(t, sums, sums + C, (long long)N * C, C);
    k_bn_fin<<<dim3(ceil_div_ll(C, 256)), dim3(256), 0, s>>>(sums, sums + C, g, b, scale, shift, N, C);
}

static void h_apply(const float* t, const float* sc, const float* sh,
                    const float* res, const float* scB, const float* shB,
                    float* o32, h16* o16, long long n, int C, hipStream_t s)
{
    k_bn_apply<<<dim3(ceil_div_ll(n, 256)), dim3(256), 0, s>>>(t, sc, sh, res, scB, shB, o32, o16, n, C - 1);
}

extern "C" void kernel_launch(void* const* d_in, const int* in_sizes, int n_in,
                              void* d_out, int out_size, void* d_ws, size_t ws_size,
                              hipStream_t stream)
{
    (void)in_sizes; (void)n_in; (void)out_size; (void)ws_size;

    // ---- input indexing (jax pytree order: sorted dict keys, tuples in order) ----
    const int* mp_batch    = (const int*)d_in[0];
    const int* mp_c1_in    = (const int*)d_in[1];
    const int* mp_c1_out   = (const int*)d_in[2];
    const int* mp_c5_in    = (const int*)d_in[3];
    const int* mp_c5_out   = (const int*)d_in[4];
    const int* mp_ds[4]    = {(const int*)d_in[5], (const int*)d_in[6],
                              (const int*)d_in[7], (const int*)d_in[8]};
    const int* mp_pool_in  = (const int*)d_in[9];
    const int* mp_pool_out = (const int*)d_in[10];
    const int* mp_s_in[4]  = {(const int*)d_in[11], (const int*)d_in[13],
                              (const int*)d_in[15], (const int*)d_in[17]};
    const int* mp_s_out[4] = {(const int*)d_in[12], (const int*)d_in[14],
                              (const int*)d_in[16], (const int*)d_in[18]};
    const int* mp_u_in[4]  = {(const int*)d_in[19], (const int*)d_in[21],
                              (const int*)d_in[23], (const int*)d_in[25]};
    const int* mp_u_out[4] = {(const int*)d_in[20], (const int*)d_in[22],
                              (const int*)d_in[24], (const int*)d_in[26]};
    const float* bn1_b   = (const float*)d_in[27];
    const float* bn1_g   = (const float*)d_in[28];
    const float* bn5_b   = (const float*)d_in[29];
    const float* bn5_g   = (const float*)d_in[30];
    const float* conv1_w = (const float*)d_in[31];
    const float* conv5_w = (const float*)d_in[32];
    const float* final_b = (const float*)d_in[33];
    const float* final_w = (const float*)d_in[34];
    struct Lay {
        const float *b1_bn1_b,*b1_bn1_g,*b1_bn2_b,*b1_bn2_g,*b1_dsbn_b,*b1_dsbn_g,*b1_dsw,*b1_w1,*b1_w2;
        const float *b2_bn1_b,*b2_bn1_g,*b2_bn2_b,*b2_bn2_g,*b2_w1,*b2_w2;
    } L[4];
    for (int i = 0; i < 4; ++i) {
        const int bse = 35 + 15 * i;
        L[i].b1_bn1_b  = (const float*)d_in[bse + 0];
        L[i].b1_bn1_g  = (const float*)d_in[bse + 1];
        L[i].b1_bn2_b  = (const float*)d_in[bse + 2];
        L[i].b1_bn2_g  = (const float*)d_in[bse + 3];
        L[i].b1_dsbn_b = (const float*)d_in[bse + 4];
        L[i].b1_dsbn_g = (const float*)d_in[bse + 5];
        L[i].b1_dsw    = (const float*)d_in[bse + 6];
        L[i].b1_w1     = (const float*)d_in[bse + 7];
        L[i].b1_w2     = (const float*)d_in[bse + 8];
        L[i].b2_bn1_b  = (const float*)d_in[bse + 9];
        L[i].b2_bn1_g  = (const float*)d_in[bse + 10];
        L[i].b2_bn2_b  = (const float*)d_in[bse + 11];
        L[i].b2_bn2_g  = (const float*)d_in[bse + 12];
        L[i].b2_w1     = (const float*)d_in[bse + 13];
        L[i].b2_w2     = (const float*)d_in[bse + 14];
    }
    const float* x = (const float*)d_in[95];

    // ---- workspace carve (~204 MB: features ~123 MB + packed f16 weights ~81 MB) ----
    uint8_t* base = (uint8_t*)d_ws;
    size_t off = 0;
    auto carve = [&](size_t bytes) -> void* {
        off = (off + 255) & ~(size_t)255;
        void* r = base + off;
        off += bytes;
        return r;
    };
    float* bufA = (float*)carve((size_t)RN_N1 * 64 * 4);     // conv1 out / f1
    float* bufB = (float*)carve((size_t)RN_N2 * 64 * 4);     // pool out / ping-pong
    float* bufC = (float*)carve((size_t)2560000 * 4);        // layer temps (max n*c)
    float* bufD = (float*)carve((size_t)2560000 * 4);
    float* bufE = (float*)carve((size_t)2560000 * 4);
    h16*   sh1  = (h16*)carve((size_t)RN_N2 * 64 * 2);       // feature f16 shadow
    h16*   sh2  = (h16*)carve((size_t)2560000 * 2);          // y f16 shadow
    h16*   x16  = (h16*)carve((size_t)RN_N0 * 32 * 2);       // x padded to 32 ch
    float* cnt  = (float*)carve((size_t)RN_N2 * 4);
    float* stat = (float*)carve((size_t)1024 * 4);
    float* scA  = (float*)carve((size_t)512 * 4);
    float* shA  = (float*)carve((size_t)512 * 4);
    float* scB  = (float*)carve((size_t)512 * 4);
    float* shB  = (float*)carve((size_t)512 * 4);
    float* gmax = (float*)carve((size_t)RN_B * 512 * 4);

    // packed weight regions + pack launches
    auto pack = [&](const float* W, int K, int Cin, int Cout) -> h16* {
        const int CinPad = (Cin + 31) & ~31;
        const long long nh = (long long)K * CinPad * Cout;
        h16* P = (h16*)carve((size_t)nh * 2);
        k_pack_w<<<dim3(ceil_div_ll(nh, 256)), dim3(256), 0, stream>>>(W, P, K, Cin, Cout, CinPad >> 5);
        return P;
    };
    h16* Pc1 = pack(conv1_w, RN_K1, RN_CIN, RN_C0);
    h16 *Pw1[4], *Pw2[4], *Pds[4], *Pb2w1[4], *Pb2w2[4];
    {
        int cin = RN_C0;
        for (int i = 0; i < 4; ++i) {
            const int c = RN_PL[i];
            Pw1[i]   = pack(L[i].b1_w1, RN_K3, cin, c);
            Pw2[i]   = pack(L[i].b1_w2, RN_K3, c, c);
            Pds[i]   = pack(L[i].b1_dsw, 1, cin, c);
            Pb2w1[i] = pack(L[i].b2_w1, RN_K3, c, c);
            Pb2w2[i] = pack(L[i].b2_w2, RN_K3, c, c);
            cin = c;
        }
    }
    h16* Pc5 = pack(conv5_w, RN_K3, 512, 512);

    // ---- pipeline ----
    k_cvt_pad<<<dim3(ceil_div_ll((long long)RN_N0 * 32, 256)), dim3(256), 0, stream>>>(
        x, x16, (long long)RN_N0 * 32);

    // conv1 + bn1 + relu (f1 fp32 in place; only pooled afterwards)
    h_conv(x16, Pc1, mp_c1_in, mp_c1_out, bufA, RN_K1, RN_M1, 32, RN_C0, RN_N1, stream);
    h_bn(bufA, bn1_g, bn1_b, RN_N1, RN_C0, stat, scA, shA, stream);
    h_apply(bufA, scA, shA, nullptr, nullptr, nullptr, bufA, nullptr,
            (long long)RN_N1 * RN_C0, RN_C0, stream);

    // avg pool -> bufB (+ sh1 shadow)
    h_zero(bufB, (long long)RN_N2 * RN_C0, stream);
    h_zero(cnt, RN_N2, stream);
    k_pool_scatter<<<dim3(ceil_div_ll((long long)4 * RN_N2 * RN_C0, 256)), dim3(256), 0, stream>>>(
        bufA, mp_pool_in, mp_pool_out, bufB, cnt, (long long)4 * RN_N2 * RN_C0, 6);
    k_pool_fin<<<dim3(ceil_div_ll((long long)RN_N2 * RN_C0, 256)), dim3(256), 0, stream>>>(
        bufB, cnt, sh1, (long long)RN_N2 * RN_C0, 6);

    int cin = RN_C0;
    for (int i = 0; i < 4; ++i) {
        const int n = RN_NL[i], c = RN_PL[i];
        const long long nc = (long long)n * c;

        // ---- basic_block_strided: in (bufB, sh1) -> out (bufC, sh1) ----
        h_conv(sh1, Pw1[i], mp_s_in[i], mp_s_out[i], bufC, RN_K3, n, cin, c, n, stream);
        h_bn(bufC, L[i].b1_bn1_g, L[i].b1_bn1_b, n, c, stat, scA, shA, stream);
        h_apply(bufC, scA, shA, nullptr, nullptr, nullptr, bufC, sh2, nc, c, stream);

        h_conv(sh2, Pw2[i], mp_u_in[i], mp_u_out[i], bufD, RN_K3, n, c, c, n, stream);
        h_conv(sh1, Pds[i], mp_ds[i], nullptr, bufE, 1, n, cin, c, n, stream);       // 1x1 ds
        h_bn(bufD, L[i].b1_bn2_g, L[i].b1_bn2_b, n, c, stat, scA, shA, stream);
        h_bn(bufE, L[i].b1_dsbn_g, L[i].b1_dsbn_b, n, c, stat, scB, shB, stream);
        h_apply(bufD, scA, shA, bufE, scB, shB, bufC, sh1, nc, c, stream);           // relu(bn+bn)

        // ---- basic_block: in (bufC, sh1) -> out (bufB, sh1) ----
        h_conv(sh1, Pb2w1[i], mp_u_in[i], mp_u_out[i], bufD, RN_K3, n, c, c, n, stream);
        h_bn(bufD, L[i].b2_bn1_g, L[i].b2_bn1_b, n, c, stat, scA, shA, stream);
        h_apply(bufD, scA, shA, nullptr, nullptr, nullptr, bufD, sh2, nc, c, stream);

        h_conv(sh2, Pb2w2[i], mp_u_in[i], mp_u_out[i], bufE, RN_K3, n, c, c, n, stream);
        h_bn(bufE, L[i].b2_bn2_g, L[i].b2_bn2_b, n, c, stat, scA, shA, stream);
        h_apply(bufE, scA, shA, bufC, nullptr, nullptr, bufB, sh1, nc, c, stream);   // relu(bn+f)

        cin = c;
    }

    // conv5 + bn5 + relu
    h_conv(sh1, Pc5, mp_c5_in, mp_c5_out, bufC, RN_K3, RN_N5, 512, 512, RN_N5, stream);
    h_bn(bufC, bn5_g, bn5_b, RN_N5, 512, stat, scA, shA, stream);
    h_apply(bufC, scA, shA, nullptr, nullptr, nullptr, bufC, nullptr,
            (long long)RN_N5 * 512, 512, stream);

    // global max pool per cloud + final FC
    h_zero(gmax, (long long)RN_B * 512, stream);
    k_seg_max<<<dim3(ceil_div_ll((long long)RN_N5 * 512, 256)), dim3(256), 0, stream>>>(
        bufC, mp_batch, gmax, (long long)RN_N5 * 512, 9);
    k_fc<<<dim3(1), dim3(256), 0, stream>>>(gmax, final_w, final_b, (float*)d_out,
                                            RN_B, 512, RN_OUT);
}
